// CausalTransformer_58445914964466
// MI455X (gfx1250) — matrix operand order repaired and checked by reading, not compiled, not run on hardware
//
#include <hip/hip_runtime.h>
#include <hip/hip_bf16.h>
#include <math.h>

// ---------------------------------------------------------------------------
// CDNA5 (gfx1250) transformer forward pass.
// All GEMMs use v_wmma_f32_16x16x32_f16 (wave32 WMMA, f32 accumulate),
// register-staged b128 global loads, software-pipelined against WMMA compute.
// ---------------------------------------------------------------------------

typedef __attribute__((ext_vector_type(16))) _Float16 v16h;
typedef __attribute__((ext_vector_type(8)))  _Float16 v8h;
typedef __attribute__((ext_vector_type(4)))  _Float16 v4h;
typedef __attribute__((ext_vector_type(8)))  float    v8f;
typedef __attribute__((ext_vector_type(4)))  float    v4f;

#define GEMM_TB 128   // 4 waves per block
#define GEMM_BM 64
#define GEMM_BN 64
#define GEMM_BK 32
#define LDS_PAD 8     // row stride = 40 halves = 80B (16B aligned)

// C[m,n] = sum_k A[m,k] * B'[k,n]  (+bias[n]) (gelu) (+Res[m,n])
// B' = B (N x K rowmajor, TRANSB=0: B[n*ldb+k])  or  B (K x N rowmajor, TRANSB=1: B[k*ldb+n])
// Batched via blockIdx.z: b = z / nH, h = z % nH; per-operand offsets b*s?b + h*s?h.
template<bool TRANSB>
__global__ __launch_bounds__(GEMM_TB) void gemm_wmma_kernel(
    const float* __restrict__ A, const float* __restrict__ B, float* __restrict__ C,
    const float* __restrict__ bias, const float* __restrict__ Res,
    int M, int N, int K, int lda, int ldb, int ldc, int geluFlag,
    int nH, long sAb, long sAh, long sBb, long sBh, long sCb, long sCh)
{
  __shared__ __align__(16) _Float16 As[GEMM_BM][GEMM_BK + LDS_PAD];
  __shared__ __align__(16) _Float16 Bs[GEMM_BN][GEMM_BK + LDS_PAD];

  const int t    = threadIdx.x;
  const int wave = t >> 5, lane = t & 31;
  const int wm   = wave >> 1, wn = wave & 1;        // 2x2 wave grid -> 32x32 per wave
  const int lr   = lane & 15, kh = lane >> 4;       // lane row/col, K-half select

  const int mBase = blockIdx.y * GEMM_BM;
  const int nBase = blockIdx.x * GEMM_BN;
  const int bz = blockIdx.z;
  const int bb = bz / nH, hh = bz - bb * nH;
  const long offA = (long)bb * sAb + (long)hh * sAh;
  const long offB = (long)bb * sBb + (long)hh * sBh;
  const long offC = (long)bb * sCb + (long)hh * sCh;

  // M and K are always full multiples in this model; only N has edge tiles.
  const bool full = (mBase + GEMM_BM <= M) && (nBase + GEMM_BN <= N);

  // ---- register staging: 4 x v4f (global_load_b128) per operand per K-step
  const int ar = t >> 3;          // row group 0..15 (A and non-trans B)
  const int ac = (t & 7) * 4;     // col within 32-wide K slab
  const int tn4 = (t & 15) * 4;   // TRANSB: n offset within tile
  const int tkr = t >> 4;         // TRANSB: k row 0..7
  v4f aR[4], bR[4];

  auto loadA = [&](int ks) {
#pragma unroll
    for (int it = 0; it < 4; ++it) {
      const int gm = mBase + ar + it * 16;
      v4f x = {0.f, 0.f, 0.f, 0.f};
      if (full || gm < M) x = *(const v4f*)(A + offA + (long)gm * lda + ks + ac);
      aR[it] = x;
    }
  };
  auto loadB = [&](int ks) {
    if (!TRANSB) {
#pragma unroll
      for (int it = 0; it < 4; ++it) {
        const int gn = nBase + ar + it * 16;
        v4f x = {0.f, 0.f, 0.f, 0.f};
        if (full || gn < N) x = *(const v4f*)(B + offB + (long)gn * ldb + ks + ac);
        bR[it] = x;
      }
    } else {
#pragma unroll
      for (int it = 0; it < 4; ++it) {
        const long kk = ks + tkr + it * 8;
        v4f x = {0.f, 0.f, 0.f, 0.f};
        if (full || (nBase + tn4 + 3) < N) {
          x = *(const v4f*)(B + offB + kk * ldb + nBase + tn4);
        } else {
#pragma unroll
          for (int j = 0; j < 4; ++j)
            if (nBase + tn4 + j < N) x[j] = B[offB + kk * ldb + nBase + tn4 + j];
        }
        bR[it] = x;
      }
    }
  };
  auto stage = [&]() {
#pragma unroll
    for (int it = 0; it < 4; ++it) {
      const v4f x = aR[it];
      const v4h h = { (_Float16)x[0], (_Float16)x[1], (_Float16)x[2], (_Float16)x[3] };
      *(v4h*)&As[ar + it * 16][ac] = h;          // ds_store_b64
    }
    if (!TRANSB) {
#pragma unroll
      for (int it = 0; it < 4; ++it) {
        const v4f x = bR[it];
        const v4h h = { (_Float16)x[0], (_Float16)x[1], (_Float16)x[2], (_Float16)x[3] };
        *(v4h*)&Bs[ar + it * 16][ac] = h;
      }
    } else {
#pragma unroll
      for (int it = 0; it < 4; ++it) {
        const v4f x = bR[it];
        const int kk = tkr + it * 8;
#pragma unroll
        for (int j = 0; j < 4; ++j) Bs[tn4 + j][kk] = (_Float16)x[j];
      }
    }
  };

  v8f acc[2][2];
#pragma unroll
  for (int mi = 0; mi < 2; ++mi)
#pragma unroll
    for (int ni = 0; ni < 2; ++ni)
      acc[mi][ni] = {};

  loadA(0);
  loadB(0);
  for (int k0 = 0; k0 < K; k0 += GEMM_BK) {
    stage();                       // waits prior loads, cvt f32->f16, ds_store_b64
    __syncthreads();
    if (k0 + GEMM_BK < K) {        // software pipeline: next tile loads overlap WMMA
      loadA(k0 + GEMM_BK);
      loadB(k0 + GEMM_BK);
    }

    // ---- fragments per ISA VGPR layouts ----
    // A 16x32 (lane m=lr): elems 0..7 = K kh*8..+7 ; elems 8..15 = K 16+kh*8..+7
    v16h af[2], bf[2];
#pragma unroll
    for (int mi = 0; mi < 2; ++mi) {
      const _Float16* p = &As[wm * 32 + mi * 16 + lr][0];
      v8h lo = *(const v8h*)(p + kh * 8);
      v8h hi = *(const v8h*)(p + 16 + kh * 8);
      af[mi] = __builtin_shufflevector(lo, hi, 0,1,2,3,4,5,6,7,8,9,10,11,12,13,14,15);
    }
    // B 32x16 (lane n=lr): elems 0..7 = K kh*8..+7 ; elems 8..15 = K 16+kh*8..+7
#pragma unroll
    for (int ni = 0; ni < 2; ++ni) {
      const _Float16* p = &Bs[wn * 32 + ni * 16 + lr][0];
      v8h lo = *(const v8h*)(p + kh * 8);
      v8h hi = *(const v8h*)(p + 16 + kh * 8);
      bf[ni] = __builtin_shufflevector(lo, hi, 0,1,2,3,4,5,6,7,8,9,10,11,12,13,14,15);
    }

#pragma unroll
    for (int mi = 0; mi < 2; ++mi)
#pragma unroll
      for (int ni = 0; ni < 2; ++ni)
        acc[mi][ni] = __builtin_amdgcn_wmma_f32_16x16x32_f16(
            false, af[mi], false, bf[ni], (short)0, acc[mi][ni], false, false);

    __syncthreads();
  }

  // ---- epilogue: bias / exact GELU / residual ----
#pragma unroll
  for (int mi = 0; mi < 2; ++mi) {
#pragma unroll
    for (int ni = 0; ni < 2; ++ni) {
#pragma unroll
      for (int r = 0; r < 8; ++r) {
        const int gm = mBase + wm * 32 + mi * 16 + r + 8 * kh;
        const int gn = nBase + wn * 32 + ni * 16 + lr;
        if (full || (gm < M && gn < N)) {
          float v = acc[mi][ni][r];
          if (bias) v += bias[gn];
          if (geluFlag) v = 0.5f * v * (1.0f + erff(v * 0.70710678118f));
          const long ci = offC + (long)gm * ldc + gn;
          if (Res) v += Res[ci];
          C[ci] = v;
        }
      }
    }
  }
}

// ---------------------------------------------------------------------------
// token embedding + sinusoidal positional encoding (computed in-kernel)
__global__ __launch_bounds__(256) void embed_kernel(
    float* __restrict__ out, const float* __restrict__ table,
    const int* __restrict__ tok, int S_, int E_)
{
  const int row = blockIdx.x;             // b*S + s
  const int s = row % S_;
  const int id = tok[row];
  const float* er = table + (long)id * E_;
  float* op = out + (long)row * E_;
  const float kLog = logf(10000.f) / (float)E_;
  for (int e = threadIdx.x; e < E_; e += blockDim.x) {
    const int epair = e & ~1;
    const float div = __expf(-(float)epair * kLog);
    const float ang = (float)s * div;
    const float pe = (e & 1) ? __cosf(ang) : __sinf(ang);
    op[e] = er[e] + pe;
  }
}

// ---------------------------------------------------------------------------
// LayerNorm over last dim (E=768; 256 threads, 3 elems each). Optional fused
// residual:  out = LN(X + R)
__global__ __launch_bounds__(256) void layernorm_kernel(
    const float* __restrict__ X, const float* __restrict__ R,
    const float* __restrict__ g, const float* __restrict__ b,
    float* __restrict__ out, int E_)
{
  const int row = blockIdx.x;
  const int t = threadIdx.x;
  const float* xp = X + (long)row * E_;
  const float* rp = R ? (R + (long)row * E_) : nullptr;
  __shared__ float red[8];

  float v[3];
  float s = 0.f;
#pragma unroll
  for (int j = 0; j < 3; ++j) {
    const int e = t + j * 256;
    float val = (e < E_) ? xp[e] : 0.f;
    if (rp && e < E_) val += rp[e];
    v[j] = val;
    s += val;
  }
  for (int o = 16; o > 0; o >>= 1) s += __shfl_xor(s, o);
  if ((t & 31) == 0) red[t >> 5] = s;
  __syncthreads();
  float tot = 0.f;
#pragma unroll
  for (int i = 0; i < 8; ++i) tot += red[i];
  const float mu = tot / (float)E_;

  float vs = 0.f;
#pragma unroll
  for (int j = 0; j < 3; ++j) {
    const int e = t + j * 256;
    if (e < E_) { const float d = v[j] - mu; vs += d * d; }
  }
  for (int o = 16; o > 0; o >>= 1) vs += __shfl_xor(vs, o);
  __syncthreads();
  if ((t & 31) == 0) red[t >> 5] = vs;
  __syncthreads();
  float vtot = 0.f;
#pragma unroll
  for (int i = 0; i < 8; ++i) vtot += red[i];
  const float rstd = rsqrtf(vtot / (float)E_ + 1e-5f);

#pragma unroll
  for (int j = 0; j < 3; ++j) {
    const int e = t + j * 256;
    if (e < E_) out[(long)row * E_ + e] = (v[j] - mu) * rstd * g[e] + b[e];
  }
}

// ---------------------------------------------------------------------------
// Row softmax with scale and optional causal mask. One block per row
// (rows = B*H*Sq, row length Skv = 512 -> 2 elems per thread @ 256 threads).
__global__ __launch_bounds__(256) void softmax_kernel(
    float* __restrict__ S, int Skv, int Sq, int causal, float scale)
{
  const int row = blockIdx.x;
  const int q = row % Sq;
  float* p = S + (long)row * Skv;
  const int valid = causal ? (q + 1) : Skv;
  const int t = threadIdx.x;
  __shared__ float red[8];

  const int i0 = t, i1 = t + 256;
  float x0 = -INFINITY, x1 = -INFINITY;
  if (i0 < valid) x0 = p[i0] * scale;
  if (i1 < valid) x1 = p[i1] * scale;

  float m = fmaxf(x0, x1);
  for (int o = 16; o > 0; o >>= 1) m = fmaxf(m, __shfl_xor(m, o));
  if ((t & 31) == 0) red[t >> 5] = m;
  __syncthreads();
  float gm = red[0];
#pragma unroll
  for (int i = 1; i < 8; ++i) gm = fmaxf(gm, red[i]);

  const float e0 = (i0 < valid) ? __expf(x0 - gm) : 0.f;
  const float e1 = (i1 < valid) ? __expf(x1 - gm) : 0.f;
  float s = e0 + e1;
  for (int o = 16; o > 0; o >>= 1) s += __shfl_xor(s, o);
  __syncthreads();
  if ((t & 31) == 0) red[t >> 5] = s;
  __syncthreads();
  float gs = 0.f;
#pragma unroll
  for (int i = 0; i < 8; ++i) gs += red[i];
  const float inv = 1.f / gs;

  if (i0 < Skv) p[i0] = e0 * inv;
  if (i1 < Skv) p[i1] = e1 * inv;
}

// ---------------------------------------------------------------------------
extern "C" void kernel_launch(void* const* d_in, const int* in_sizes, int n_in,
                              void* d_out, int out_size, void* d_ws, size_t ws_size,
                              hipStream_t stream) {
  (void)in_sizes; (void)n_in; (void)out_size; (void)ws_size;

  constexpr int E = 768, H = 8, DH = 96, F = 2048, V = 30522, NL = 6, S = 512, B = 2;
  constexpr int M = B * S;  // 1024 token rows

  struct MHAp { const float *Wq, *Wk, *Wv, *Wo, *bo; };
  struct LNp  { const float *g, *b; };
  struct FFp  { const float *W1, *b1, *W2, *b2; };

  int p = 0;
  auto nf = [&]() { return (const float*)d_in[p++]; };
  const int* src = (const int*)d_in[p++];
  const int* trg = (const int*)d_in[p++];
  const float* emb_src = nf();
  const float* emb_trg = nf();

  MHAp encA[NL]; LNp encL1[NL]; FFp encF[NL]; LNp encL2[NL];
  for (int l = 0; l < NL; ++l) {
    encA[l]  = { nf(), nf(), nf(), nf(), nf() };
    encL1[l] = { nf(), nf() };
    encF[l]  = { nf(), nf(), nf(), nf() };
    encL2[l] = { nf(), nf() };
  }
  MHAp decS[NL], decC[NL]; LNp dL1[NL], dL2[NL], dL3[NL]; FFp decF[NL];
  for (int l = 0; l < NL; ++l) {
    decS[l] = { nf(), nf(), nf(), nf(), nf() };
    decC[l] = { nf(), nf(), nf(), nf(), nf() };
    dL1[l]  = { nf(), nf() };
    dL2[l]  = { nf(), nf() };
    dL3[l]  = { nf(), nf() };
    decF[l] = { nf(), nf(), nf(), nf() };
  }
  LNp lnf = { nf(), nf() };
  const float* Wlm = nf();
  const float* blm = nf();

  // ---- workspace carve-out (floats) ----
  float* w = (float*)d_ws;
  size_t o = 0;
  auto alloc = [&](size_t n) { float* r = w + o; o += n; return r; };
  float* X   = alloc((size_t)M * E);        // encoder state / enc_out
  float* Hh  = alloc((size_t)M * E);        // encoder post-attn LN
  float* T   = alloc((size_t)M * E);        // temp
  float* Q   = alloc((size_t)M * E);
  float* Kb  = alloc((size_t)M * E);
  float* Vb  = alloc((size_t)M * E);
  float* Cc  = alloc((size_t)M * E);        // attn output, heads concatenated
  float* Y   = alloc((size_t)M * E);        // decoder state
  float* Nn  = alloc((size_t)M * E);        // decoder pre-norm
  float* FF1 = alloc((size_t)M * F);
  float* Sc  = alloc((size_t)B * H * S * S);// attention scores

  const float scale = 0.1020620726f;        // 1/sqrt(96)

  auto gemmB = [&](const float* A, const float* Bp, float* C, const float* bias,
                   const float* res, int M_, int N_, int K_, int lda, int ldb, int ldc,
                   bool transB, int gelu, int batches, int nH,
                   long sAb, long sAh, long sBb, long sBh, long sCb, long sCh) {
    dim3 grid((N_ + GEMM_BN - 1) / GEMM_BN, (M_ + GEMM_BM - 1) / GEMM_BM, batches);
    if (transB)
      gemm_wmma_kernel<true><<<grid, GEMM_TB, 0, stream>>>(
          A, Bp, C, bias, res, M_, N_, K_, lda, ldb, ldc, gelu,
          nH, sAb, sAh, sBb, sBh, sCb, sCh);
    else
      gemm_wmma_kernel<false><<<grid, GEMM_TB, 0, stream>>>(
          A, Bp, C, bias, res, M_, N_, K_, lda, ldb, ldc, gelu,
          nH, sAb, sAh, sBb, sBh, sCb, sCh);
  };
  auto gemm1 = [&](const float* A, const float* Bp, float* C, const float* bias,
                   const float* res, int M_, int N_, int K_, int gelu) {
    gemmB(A, Bp, C, bias, res, M_, N_, K_, K_, K_, N_, false, gelu,
          1, 1, 0, 0, 0, 0, 0, 0);
  };
  // batched multi-head attention core on (q,k,v) buffers laid out [B,S,H*DH]
  auto attention = [&](const float* qb, const float* kb, const float* vb,
                       float* outCc, int causal) {
    // scores[b,h] = Q K^T  (M=S, N=S, K=DH)
    gemmB(qb, kb, Sc, nullptr, nullptr, S, S, DH, E, E, S, false, 0,
          B * H, H,
          (long)S * E, (long)DH,            // A offsets (b,h)
          (long)S * E, (long)DH,            // B offsets (b,h)
          (long)H * S * S, (long)S * S);    // C offsets (b,h)
    softmax_kernel<<<B * H * S, 256, 0, stream>>>(Sc, S, S, causal, scale);
    // out[b,h] = attn V  (M=S, N=DH, K=S), V accessed transposed (K x N)
    gemmB(Sc, vb, outCc, nullptr, nullptr, S, DH, S, S, E, E, true, 0,
          B * H, H,
          (long)H * S * S, (long)S * S,
          (long)S * E, (long)DH,
          (long)S * E, (long)DH);           // heads land concatenated in ldc=E
  };

  // ================= embeddings =================
  embed_kernel<<<M, 256, 0, stream>>>(X, emb_src, src, S, E);
  embed_kernel<<<M, 256, 0, stream>>>(Y, emb_trg, trg, S, E);

  // ================= encoder (post-norm) =================
  for (int l = 0; l < NL; ++l) {
    gemm1(X, encA[l].Wq, Q,  nullptr, nullptr, M, E, E, 0);
    gemm1(X, encA[l].Wk, Kb, nullptr, nullptr, M, E, E, 0);
    gemm1(X, encA[l].Wv, Vb, nullptr, nullptr, M, E, E, 0);
    attention(Q, Kb, Vb, Cc, /*causal=*/0);
    gemm1(Cc, encA[l].Wo, T, encA[l].bo, nullptr, M, E, E, 0);
    layernorm_kernel<<<M, 256, 0, stream>>>(X, T, encL1[l].g, encL1[l].b, Hh, E);
    gemm1(Hh, encF[l].W1, FF1, encF[l].b1, nullptr, M, F, E, /*gelu=*/1);
    gemm1(FF1, encF[l].W2, T, encF[l].b2, Hh /*residual*/, M, E, F, 0);
    layernorm_kernel<<<M, 256, 0, stream>>>(T, nullptr, encL2[l].g, encL2[l].b, X, E);
  }
  // X now holds enc_out.

  // ================= decoder (pre-norm) =================
  for (int l = 0; l < NL; ++l) {
    // self-attention (causal)
    layernorm_kernel<<<M, 256, 0, stream>>>(Y, nullptr, dL1[l].g, dL1[l].b, Nn, E);
    gemm1(Nn, decS[l].Wq, Q,  nullptr, nullptr, M, E, E, 0);
    gemm1(Nn, decS[l].Wk, Kb, nullptr, nullptr, M, E, E, 0);
    gemm1(Nn, decS[l].Wv, Vb, nullptr, nullptr, M, E, E, 0);
    attention(Q, Kb, Vb, Cc, /*causal=*/1);
    gemm1(Cc, decS[l].Wo, Y, decS[l].bo, Y /*residual in-place*/, M, E, E, 0);
    // cross-attention (kv from enc_out)
    layernorm_kernel<<<M, 256, 0, stream>>>(Y, nullptr, dL2[l].g, dL2[l].b, Nn, E);
    gemm1(Nn, decC[l].Wq, Q,  nullptr, nullptr, M, E, E, 0);
    gemm1(X,  decC[l].Wk, Kb, nullptr, nullptr, M, E, E, 0);
    gemm1(X,  decC[l].Wv, Vb, nullptr, nullptr, M, E, E, 0);
    attention(Q, Kb, Vb, Cc, /*causal=*/0);
    gemm1(Cc, decC[l].Wo, Y, decC[l].bo, Y, M, E, E, 0);
    // feed-forward
    layernorm_kernel<<<M, 256, 0, stream>>>(Y, nullptr, dL3[l].g, dL3[l].b, Nn, E);
    gemm1(Nn, decF[l].W1, FF1, decF[l].b1, nullptr, M, F, E, /*gelu=*/1);
    gemm1(FF1, decF[l].W2, Y, decF[l].b2, Y, M, E, F, 0);
  }

  // ================= final LN + LM head =================
  layernorm_kernel<<<M, 256, 0, stream>>>(Y, nullptr, lnf.g, lnf.b, Nn, E);
  gemm1(Nn, Wlm, (float*)d_out, blm, nullptr, M, V, E, 0);
}
